// LocalWindowAttentionBlock_57801669869934
// MI455X (gfx1250) — matrix operand order repaired and checked by reading, not compiled, not run on hardware
//
#include <hip/hip_runtime.h>
#include <math.h>

typedef __attribute__((ext_vector_type(16))) _Float16 v16h;
typedef __attribute__((ext_vector_type(8)))  _Float16 v8h;
typedef __attribute__((ext_vector_type(8)))  float    v8f;

#define BB   4
#define LL   2048
#define DD   512
#define HH   8
#define HD   64
#define FF_  2048
#define WIN  4
#define ROWS (BB * LL)          // 8192 token rows

#define TM   128                // block tile M
#define TN   128                // block tile N
#define TK   32                 // K-step (one 16x16x32 WMMA depth)
#define LPAD 40                 // LDS row pitch in halfs: 80B -> 16B aligned, bank stride 20

// ---------------------------------------------------------------------------
// CDNA5 async global->LDS copy (ASYNCcnt) — inline asm, per CDNA5 ISA
// ---------------------------------------------------------------------------
__device__ __forceinline__ void async_b128(_Float16* lds, const _Float16* g) {
    uint32_t l = (uint32_t)(uintptr_t)lds;   // low 32 bits = LDS offset
    asm volatile("global_load_async_to_lds_b128 %0, %1, off"
                 :: "v"(l), "v"(g) : "memory");
}
__device__ __forceinline__ void wait_async0() {
    asm volatile("s_wait_asynccnt 0x0" ::: "memory");
}

// ---------------------------------------------------------------------------
// f32 -> f16 elementwise cast (weights)
// ---------------------------------------------------------------------------
__global__ __launch_bounds__(256)
void cast_f16_kernel(const float* __restrict__ src, _Float16* __restrict__ dst, int n) {
    int i = blockIdx.x * 256 + threadIdx.x;
    if (i < n) dst[i] = (_Float16)src[i];
}

// ---------------------------------------------------------------------------
// LayerNorm over D=512, one wave32 per row, 8 rows per 256-thread block.
// ---------------------------------------------------------------------------
__global__ __launch_bounds__(256)
void ln_f16_kernel(const float* __restrict__ x, const float* __restrict__ g,
                   const float* __restrict__ b, _Float16* __restrict__ out) {
    const int lane = threadIdx.x & 31;
    const int wave = threadIdx.x >> 5;
    const int row  = blockIdx.x * 8 + wave;
    const float* xr = x + (size_t)row * DD;

    float vals[16];
    float s = 0.f;
#pragma unroll
    for (int i = 0; i < 16; ++i) { vals[i] = xr[lane * 16 + i]; s += vals[i]; }
#pragma unroll
    for (int off = 16; off >= 1; off >>= 1) s += __shfl_xor(s, off, 32);
    const float mean = s * (1.f / DD);

    float v = 0.f;
#pragma unroll
    for (int i = 0; i < 16; ++i) { float d = vals[i] - mean; v += d * d; }
#pragma unroll
    for (int off = 16; off >= 1; off >>= 1) v += __shfl_xor(v, off, 32);
    const float rstd = rsqrtf(v * (1.f / DD) + 1e-5f);

    _Float16* orow = out + (size_t)row * DD;
#pragma unroll
    for (int i = 0; i < 16; ++i) {
        int c = lane * 16 + i;
        orow[c] = (_Float16)((vals[i] - mean) * rstd * g[c] + b[c]);
    }
}

// ---------------------------------------------------------------------------
// One K-step of the wave's 32x64 tile: preload 2 A-frags + 4 B-frags from LDS,
// then 8 back-to-back WMMAs (A-frag reused across the 4 N-subtiles).
// ---------------------------------------------------------------------------
__device__ __forceinline__ void gemm_step(const _Float16* __restrict__ aT,
                                          const _Float16* __restrict__ bT,
                                          int wm, int wn, int lmod, int half_sel,
                                          v8f acc[2][4]) {
    v16h af[2];
#pragma unroll
    for (int ii = 0; ii < 2; ++ii) {
        const _Float16* ap = aT + (wm * 32 + ii * 16 + lmod) * LPAD + half_sel * 8;
        v8h lo = *(const v8h*)(ap);
        v8h hi = *(const v8h*)(ap + 16);
#pragma unroll
        for (int e = 0; e < 8; ++e) { af[ii][e] = lo[e]; af[ii][8 + e] = hi[e]; }
    }
    v16h bf[4];
#pragma unroll
    for (int s = 0; s < 4; ++s) {
        const _Float16* bp = bT + (wn * 64 + s * 16 + lmod) * LPAD + half_sel * 8;
        v8h blo = *(const v8h*)(bp);
        v8h bhi = *(const v8h*)(bp + 16);
#pragma unroll
        for (int e = 0; e < 8; ++e) { bf[s][e] = blo[e]; bf[s][8 + e] = bhi[e]; }
    }
#pragma unroll
    for (int s = 0; s < 4; ++s) {
        acc[0][s] = __builtin_amdgcn_wmma_f32_16x16x32_f16(
                        false, af[0], false, bf[s], (short)0, acc[0][s], false, false);
        acc[1][s] = __builtin_amdgcn_wmma_f32_16x16x32_f16(
                        false, af[1], false, bf[s], (short)0, acc[1][s], false, false);
    }
}

// ---------------------------------------------------------------------------
// LDS-staged, async double-buffered WMMA GEMM:
//   out[M,N] = epilogue( A[M,K](f16) @ W[N,K]^T(f16) + bias[N] )
// 256 threads = 8 waves; block tile 128x128; wave tile 32x64.
// K-loop unrolled by 2 (nk always even) -> static buffer addressing.
// MODE 0: +bias -> f16 | MODE 1: +bias,+erf-GELU -> f16 | MODE 2: +bias,+residual -> f32
// ---------------------------------------------------------------------------
template <int MODE>
__global__ __launch_bounds__(256)
void wmma_gemm_lds_kernel(const _Float16* __restrict__ A,
                          const _Float16* __restrict__ W,
                          const float*    __restrict__ bias,
                          const float*    __restrict__ resid,
                          float*          __restrict__ outF,
                          _Float16*       __restrict__ outH,
                          int M, int N, int K) {
    __shared__ _Float16 sA[2][TM * LPAD];   // 2 x 10 KB
    __shared__ _Float16 sB[2][TN * LPAD];   // 2 x 10 KB

    const int tid      = threadIdx.x;
    const int lane     = tid & 31;
    const int wave     = tid >> 5;
    const int half_sel = lane >> 4;          // lane group 0..15 / 16..31
    const int lmod     = lane & 15;
    const int wm       = wave >> 1;          // 0..3 -> 32-row strip
    const int wn       = wave & 1;           // 0..1 -> 64-col strip

    const int m0 = blockIdx.x * TM;
    const int n0 = blockIdx.y * TN;

    // cooperative tile fill: thread -> (row, 16-half chunk)
    const int ldr = tid >> 1;                // 0..127
    const int ldc = (tid & 1) * 16;          // 0 / 16
    const _Float16* gA = A + (size_t)(m0 + ldr) * K + ldc;
    const _Float16* gB = W + (size_t)(n0 + ldr) * K + ldc;
    _Float16* lA0 = &sA[0][ldr * LPAD + ldc];
    _Float16* lA1 = &sA[1][ldr * LPAD + ldc];
    _Float16* lB0 = &sB[0][ldr * LPAD + ldc];
    _Float16* lB1 = &sB[1][ldr * LPAD + ldc];

    v8f acc[2][4];
#pragma unroll
    for (int i = 0; i < 2; ++i)
#pragma unroll
        for (int s = 0; s < 4; ++s)
            acc[i][s] = (v8f){0.f,0.f,0.f,0.f,0.f,0.f,0.f,0.f};

    // prime tile 0 -> buf 0
    async_b128(lA0,     gA);
    async_b128(lA0 + 8, gA + 8);
    async_b128(lB0,     gB);
    async_b128(lB0 + 8, gB + 8);

    const int nk = K / TK;                   // 16 or 64: always even
    for (int i = 0; i < nk; i += 2) {
        // ---- tile i (buf 0) ----
        wait_async0();                       // this wave's copies landed
        __syncthreads();                     // all waves' copies landed; prev reads done
        {                                    // stream tile i+1 -> buf 1
            const int kb = (i + 1) * TK;
            async_b128(lA1,     gA + kb);
            async_b128(lA1 + 8, gA + kb + 8);
            async_b128(lB1,     gB + kb);
            async_b128(lB1 + 8, gB + kb + 8);
        }
        gemm_step(&sA[0][0], &sB[0][0], wm, wn, lmod, half_sel, acc);

        // ---- tile i+1 (buf 1) ----
        wait_async0();
        __syncthreads();
        if (i + 2 < nk) {                    // stream tile i+2 -> buf 0
            const int kb = (i + 2) * TK;
            async_b128(lA0,     gA + kb);
            async_b128(lA0 + 8, gA + kb + 8);
            async_b128(lB0,     gB + kb);
            async_b128(lB0 + 8, gB + kb + 8);
        }
        gemm_step(&sA[1][0], &sB[1][0], wm, wn, lmod, half_sel, acc);
    }

    // epilogue: VGPR r -> (M = 16*tile + hs*8 + r, N = 16*tile + lmod)
#pragma unroll
    for (int ii = 0; ii < 2; ++ii) {
#pragma unroll
        for (int s = 0; s < 4; ++s) {
            const int n  = n0 + wn * 64 + s * 16 + lmod;
            const float bn = bias[n];
#pragma unroll
            for (int r = 0; r < 8; ++r) {
                const int m = m0 + wm * 32 + ii * 16 + half_sel * 8 + r;
                float v = acc[ii][s][r] + bn;
                if (MODE == 1) v = 0.5f * v * (1.f + erff(v * 0.70710678118654752f));
                if (MODE == 2) {
                    outF[(size_t)m * N + n] = v + resid[(size_t)m * N + n];
                } else {
                    outH[(size_t)m * N + n] = (_Float16)v;
                }
            }
        }
    }
}

// ---------------------------------------------------------------------------
// Sliding-window attention, one wave32 per (row, head). hd=64 -> 2 dims/lane.
// ---------------------------------------------------------------------------
__global__ __launch_bounds__(256)
void attn_kernel(const _Float16* __restrict__ qkv, _Float16* __restrict__ o16) {
    const int lane = threadIdx.x & 31;
    const int wave = threadIdx.x >> 5;
    const int gid  = blockIdx.x * 8 + wave;      // 0 .. ROWS*H-1
    const int h    = gid & (HH - 1);
    const int row  = gid >> 3;                    // b*L + l
    const int l    = row & (LL - 1);
    const int d    = h * HD + lane * 2;

    const _Float16* qp = qkv + (size_t)row * (3 * DD) + d;
    const float q0 = (float)qp[0];
    const float q1 = (float)qp[1];

    float sc[2 * WIN + 1];
    float mx = -1e30f;
#pragma unroll
    for (int t = 0; t < 2 * WIN + 1; ++t) {
        const int j  = l - WIN + t;
        const bool ok = (j >= 0) && (j < LL);
        const int jd = ok ? (j - l) : 0;
        const _Float16* kp = qkv + (size_t)(row + jd) * (3 * DD) + DD + d;
        float s = q0 * (float)kp[0] + q1 * (float)kp[1];
#pragma unroll
        for (int off = 16; off >= 1; off >>= 1) s += __shfl_xor(s, off, 32);
        s = ok ? (s * 0.125f) : -1e30f;           // 1/sqrt(64)
        sc[t] = s;
        mx = fmaxf(mx, s);
    }

    float denom = 0.f, o0 = 0.f, o1 = 0.f;
#pragma unroll
    for (int t = 0; t < 2 * WIN + 1; ++t) {
        const int j  = l - WIN + t;
        const bool ok = (j >= 0) && (j < LL);
        const int jd = ok ? (j - l) : 0;
        const float p = ok ? __expf(sc[t] - mx) : 0.f;
        denom += p;
        const _Float16* vp = qkv + (size_t)(row + jd) * (3 * DD) + 2 * DD + d;
        o0 += p * (float)vp[0];
        o1 += p * (float)vp[1];
    }
    const float inv = 1.f / denom;
    _Float16* op = o16 + (size_t)row * DD + d;
    op[0] = (_Float16)(o0 * inv);
    op[1] = (_Float16)(o1 * inv);
}

// ---------------------------------------------------------------------------
// Launch
// ---------------------------------------------------------------------------
extern "C" void kernel_launch(void* const* d_in, const int* in_sizes, int n_in,
                              void* d_out, int out_size, void* d_ws, size_t ws_size,
                              hipStream_t stream) {
    const float* x         = (const float*)d_in[0];
    const float* in_proj_w = (const float*)d_in[1];
    const float* in_proj_b = (const float*)d_in[2];
    const float* out_w     = (const float*)d_in[3];
    const float* out_b     = (const float*)d_in[4];
    const float* ln1_w     = (const float*)d_in[5];
    const float* ln1_b     = (const float*)d_in[6];
    const float* ln2_w     = (const float*)d_in[7];
    const float* ln2_b     = (const float*)d_in[8];
    const float* ffn_w1    = (const float*)d_in[9];
    const float* ffn_b1    = (const float*)d_in[10];
    const float* ffn_w2    = (const float*)d_in[11];
    const float* ffn_b2    = (const float*)d_in[12];
    float* out = (float*)d_out;

    // workspace carve-up
    char* p = (char*)d_ws;
    _Float16* h16    = (_Float16*)p; p += (size_t)ROWS * DD * 2;        // LN1/LN2 out (reused)
    _Float16* qkv16  = (_Float16*)p; p += (size_t)ROWS * 3 * DD * 2;    // QKV
    _Float16* o16    = (_Float16*)p; p += (size_t)ROWS * DD * 2;        // attn out
    float*    x2     = (float*)p;    p += (size_t)ROWS * DD * 4;        // residual stream 2
    _Float16* a1     = (_Float16*)p; p += (size_t)ROWS * FF_ * 2;       // FFN hidden
    _Float16* wIn16  = (_Float16*)p; p += (size_t)3 * DD * DD * 2;
    _Float16* wOut16 = (_Float16*)p; p += (size_t)DD * DD * 2;
    _Float16* w116   = (_Float16*)p; p += (size_t)FF_ * DD * 2;
    _Float16* w216   = (_Float16*)p; p += (size_t)FF_ * DD * 2;

    // weight casts
    cast_f16_kernel<<<(3 * DD * DD + 255) / 256, 256, 0, stream>>>(in_proj_w, wIn16, 3 * DD * DD);
    cast_f16_kernel<<<(DD * DD + 255) / 256, 256, 0, stream>>>(out_w, wOut16, DD * DD);
    cast_f16_kernel<<<(FF_ * DD + 255) / 256, 256, 0, stream>>>(ffn_w1, w116, FF_ * DD);
    cast_f16_kernel<<<(FF_ * DD + 255) / 256, 256, 0, stream>>>(ffn_w2, w216, FF_ * DD);

    // LN1 -> h16
    ln_f16_kernel<<<ROWS / 8, 256, 0, stream>>>(x, ln1_w, ln1_b, h16);

    // QKV GEMM: [8192,512] x [1536,512]^T -> qkv16 f16
    wmma_gemm_lds_kernel<0><<<dim3(ROWS / TM, (3 * DD) / TN), 256, 0, stream>>>(
        h16, wIn16, in_proj_b, nullptr, nullptr, qkv16, ROWS, 3 * DD, DD);

    // sliding-window attention -> o16 f16
    attn_kernel<<<(ROWS * HH) / 8, 256, 0, stream>>>(qkv16, o16);

    // out-proj + residual(x): x2 = x + o16 @ out_w^T + out_b   (f32)
    wmma_gemm_lds_kernel<2><<<dim3(ROWS / TM, DD / TN), 256, 0, stream>>>(
        o16, wOut16, out_b, x, x2, nullptr, ROWS, DD, DD);

    // LN2 -> h16 (reuse)
    ln_f16_kernel<<<ROWS / 8, 256, 0, stream>>>(x2, ln2_w, ln2_b, h16);

    // FFN1 + GELU: [8192,512] x [2048,512]^T -> a1 f16
    wmma_gemm_lds_kernel<1><<<dim3(ROWS / TM, FF_ / TN), 256, 0, stream>>>(
        h16, w116, ffn_b1, nullptr, nullptr, a1, ROWS, FF_, DD);

    // FFN2 + residual(x2): out = x2 + a1 @ w2^T + b2   (f32, K=2048)
    wmma_gemm_lds_kernel<2><<<dim3(ROWS / TM, DD / TN), 256, 0, stream>>>(
        a1, w216, ffn_b2, x2, out, nullptr, ROWS, DD, FF_);
}
